// CustomRNN_73529840107711
// MI455X (gfx1250) — matrix operand order repaired and checked by reading, not compiled, not run on hardware
//
#include <hip/hip_runtime.h>
#include <hip/hip_bf16.h>

typedef __attribute__((ext_vector_type(16))) _Float16 v16h;
typedef __attribute__((ext_vector_type(8)))  float    v8f;
typedef __attribute__((ext_vector_type(2)))  __fp16   fp16x2;

union H2U { fp16x2 h; unsigned int u; };
union BFr { v16h v; unsigned int w[8]; };

#define RNN_H 64

__device__ __forceinline__ float fast_tanh(float x) {
#if __has_builtin(__builtin_amdgcn_tanhf)
  return __builtin_amdgcn_tanhf(x);
#else
  // tanh(x) = 1 - 2/(1 + e^{2x}); saturates cleanly at +/-1 for large |x|
  float e = __builtin_amdgcn_exp2f(x * 2.8853900817779268f); // 2*log2(e)
  return 1.0f - 2.0f * __builtin_amdgcn_rcpf(1.0f + e);
#endif
}

// Exchange a dword with lane ^ 16 (group-of-32 swizzle: xor=0x10, and=0x1f)
__device__ __forceinline__ unsigned swap16(unsigned v) {
  return (unsigned)__builtin_amdgcn_ds_swizzle((int)v, 0x401F);
}

__global__ __launch_bounds__(256) void rnn_wmma_kernel(
    const float* __restrict__ x,   const float* __restrict__ Wih,
    const float* __restrict__ Whh, const float* __restrict__ bih,
    const float* __restrict__ bhh, const float* __restrict__ fcw,
    const float* __restrict__ fcb, float* __restrict__ out,
    int Bsz, int T)
{
  const int lane = threadIdx.x & 31;
  const int hi   = lane >> 4;     // which half-wave
  const int m    = lane & 15;     // batch column this lane owns (in D/B layouts)
  const int gwave = (blockIdx.x * blockDim.x + threadIdx.x) >> 5;
  const int base  = gwave * 16;   // wave-uniform
  if (base >= Bsz) return;        // uniform branch: EXEC stays all-1s for WMMA

  // ---- A fragments: W_hh, f16, resident in VGPRs for the whole T loop ----
  // A tile (jt,kc) covers rows j = jt*16+M, cols k = kc*32+K of W_hh.
  // A 16x32 k order here (contiguous, same as B): lane<16: element s holds K = s (0..15)
  //                        lane>=16: K = 16 + s (16..31).
  v16h afrag[4][2];
  #pragma unroll
  for (int jt = 0; jt < 4; ++jt) {
    const float* wrow = Whh + (jt * 16 + m) * RNN_H;
    #pragma unroll
    for (int kc = 0; kc < 2; ++kc) {
      #pragma unroll
      for (int v = 0; v < 8; ++v) {
        int kb = 16 * hi + 2 * v;
        afrag[jt][kc][2 * v + 0] = (_Float16)wrow[kc * 32 + kb + 0];
        afrag[jt][kc][2 * v + 1] = (_Float16)wrow[kc * 32 + kb + 1];
      }
    }
  }

  // Per-lane hidden indices in the D layout: j = jt*16 + 8*hi + r  (r = VGPR row)
  float wihv[4][8], biasv[4][8];
  #pragma unroll
  for (int jt = 0; jt < 4; ++jt)
    #pragma unroll
    for (int r = 0; r < 8; ++r) {
      int j = jt * 16 + 8 * hi + r;
      wihv[jt][r]  = Wih[j];            // I == 1
      biasv[jt][r] = bih[j] + bhh[j];
    }

  // This lane streams x for its batch row; 16B-aligned (T*4 bytes per row).
  const float4* xr = reinterpret_cast<const float4*>(x + (size_t)(base + m) * T);

  // B fragments = h^T (f16); h0 = 0.
  BFr bfr[2];
  #pragma unroll
  for (int kc = 0; kc < 2; ++kc)
    #pragma unroll
    for (int i = 0; i < 8; ++i) bfr[kc].w[i] = 0u;

  float ht[4][8];   // tanh outputs (f32) of the current step, D layout

  const int T4 = T >> 2;
  float4 xq = xr[0];                    // software-pipelined x stream
  for (int tt = 0; tt < T4; ++tt) {
    float xs[4] = {xq.x, xq.y, xq.z, xq.w};
    if (tt + 1 < T4) xq = xr[tt + 1];   // prefetch next 4 steps behind the WMMA work
    #pragma unroll
    for (int s = 0; s < 4; ++s) {
      const float xv = xs[s];

      // Phase 1: build all four C fragments (input projection + biases, folded
      // into the MMA accumulator for free).
      v8f c[4];
      #pragma unroll
      for (int jt = 0; jt < 4; ++jt)
        #pragma unroll
        for (int r = 0; r < 8; ++r)
          c[jt][r] = __builtin_fmaf(xv, wihv[jt][r], biasv[jt][r]);

      // Phase 2: all 8 WMMAs. kc-outer so consecutive WMMAs share the B
      // operand; jt-inner gives 4 independent accumulation chains that fill
      // each other's result-hazard windows.
      #pragma unroll
      for (int kc = 0; kc < 2; ++kc)
        #pragma unroll
        for (int jt = 0; jt < 4; ++jt)
          c[jt] = __builtin_amdgcn_wmma_f32_16x16x32_f16(
                      false, afrag[jt][kc], false, bfr[kc].v,
                      (short)0, c[jt], false, false);

      // Phase 3: elementwise tanh (native v_tanh_f32) + f16 pack.
      unsigned p[4][4];   // packed f16 pairs of h_t, per D tile
      #pragma unroll
      for (int jt = 0; jt < 4; ++jt) {
        #pragma unroll
        for (int r = 0; r < 8; ++r) ht[jt][r] = fast_tanh(c[jt][r]);
        #pragma unroll
        for (int i = 0; i < 4; ++i) {
          H2U u; u.h = __builtin_amdgcn_cvt_pkrtz(ht[jt][2 * i], ht[jt][2 * i + 1]);
          p[jt][i] = u.u;
        }
      }

      // Phase 4: rebuild B (h^T) fragments from D-layout results.
      // B k-chunk kc word v (lane<16):  K=2v..    -> j=32kc+2v    : tile 2kc
      //            word v (lane>=16):   K=16+2v.. -> j=32kc+16+2v : tile 2kc+1
      // Rows 0..7 of a tile live in lanes<16, rows 8..15 in lanes>=16 -> lane^16.
      #pragma unroll
      for (int kc = 0; kc < 2; ++kc) {
        #pragma unroll
        for (int i = 0; i < 4; ++i) {
          unsigned own_lo = p[2 * kc][i];
          unsigned own_hi = p[2 * kc + 1][i];
          unsigned sw_lo  = swap16(own_lo);
          unsigned sw_hi  = swap16(own_hi);
          bfr[kc].w[i]     = (hi == 0) ? own_lo : sw_hi;
          bfr[kc].w[4 + i] = (hi == 0) ? sw_lo  : own_hi;
        }
      }
    }
  }

  // ---- fc head on h_T: out[b] = sum_j h[b][j]*fcw[j] + fcb ----
  float partial = 0.0f;
  #pragma unroll
  for (int jt = 0; jt < 4; ++jt)
    #pragma unroll
    for (int r = 0; r < 8; ++r)
      partial = __builtin_fmaf(ht[jt][r], fcw[jt * 16 + 8 * hi + r], partial);
  // lane c holds j = {0..7,16..23,...}, lane c+16 the complementary j set
  unsigned pb = __float_as_uint(partial);
  float other = __uint_as_float(swap16(pb));
  if (hi == 0) out[base + m] = partial + other + fcb[0];
}

extern "C" void kernel_launch(void* const* d_in, const int* in_sizes, int n_in,
                              void* d_out, int out_size, void* d_ws, size_t ws_size,
                              hipStream_t stream) {
  const float* x   = (const float*)d_in[0];
  const float* Wih = (const float*)d_in[1];
  const float* Whh = (const float*)d_in[2];
  const float* bih = (const float*)d_in[3];
  const float* bhh = (const float*)d_in[4];
  const float* fcw = (const float*)d_in[5];
  const float* fcb = (const float*)d_in[6];
  float* out = (float*)d_out;

  const int Bsz = out_size;              // 2048
  const int T   = in_sizes[0] / Bsz;     // 512 (I == 1)

  const int waves   = (Bsz + 15) / 16;   // one wave per 16 batch rows
  const int threads = 256;               // 8 waves / workgroup
  const int blocks  = (waves * 32 + threads - 1) / threads;
  rnn_wmma_kernel<<<blocks, threads, 0, stream>>>(x, Wih, Whh, bih, bhh, fcw, fcb,
                                                  out, Bsz, T);
}
